// RPNProposalGenerator_56590489092314
// MI455X (gfx1250) — compile-verified, matrix operand order checked by reading, not run on hardware
//
#include <hip/hip_runtime.h>
#include <stdint.h>

// Problem constants (match reference)
#define A_N 100000
#define M_N 40
#define B_N 8
#define IMG_F 800.0f
#define POS_TH 0.7f
#define NEG_TH 0.3f
#define MAX_SAMPLES 256
#define MAX_POSK 128

typedef __attribute__((ext_vector_type(16))) _Float16 v16h;
typedef __attribute__((ext_vector_type(8)))  float    v8f;

// ---- monotone float <-> u32 key (total order incl. negatives) -------------
__device__ __forceinline__ uint32_t f2key(float f) {
  uint32_t u = __float_as_uint(f);
  return (u & 0x80000000u) ? ~u : (u | 0x80000000u);
}
__device__ __forceinline__ float key2f(uint32_t k) {
  uint32_t u = (k & 0x80000000u) ? (k & 0x7fffffffu) : ~k;
  return __uint_as_float(u);
}

__device__ __forceinline__ uint32_t hash32(uint32_t x) {
  x ^= x >> 16; x *= 0x7feb352du;
  x ^= x >> 15; x *= 0x846ca68bu;
  x ^= x >> 16; return x;
}

// ---- gfx1250 async global->LDS staging (ASYNCcnt path) --------------------
__device__ __forceinline__ void async_b128_to_lds(uint32_t lds_byte_addr, const void* gaddr) {
  asm volatile("global_load_async_to_lds_b128 %0, %1, off"
               :: "v"(lds_byte_addr), "v"(gaddr) : "memory");
}
__device__ __forceinline__ void wait_async0() {
  asm volatile("s_wait_asynccnt 0" ::: "memory");
}

// masked IoU of anchor av (precomputed area_a) vs gt g
__device__ __forceinline__ float iou_masked(float4 av, float area_a, float4 g, bool inimg) {
  float lx = fmaxf(av.x, g.x), ly = fmaxf(av.y, g.y);
  float rx = fminf(av.z, g.z), ry = fminf(av.w, g.w);
  float w = fmaxf(rx - lx, 0.0f), h = fmaxf(ry - ly, 0.0f);
  float inter = w * h;
  float area_g = (g.z - g.x) * (g.w - g.y);
  float i = inter / (area_a + area_g - inter);
  return inimg ? i : -1.0f;
}

__device__ __forceinline__ bool anchor_inimg(float4 av) {
  bool c0 = (av.x >= 0.0f) & (av.y >= 0.0f) & (av.x <= IMG_F) & (av.y <= IMG_F);
  bool c1 = (av.z >= 0.0f) & (av.w >= 0.0f) & (av.z <= IMG_F) & (av.w <= IMG_F);
  return c0 | c1;
}

// ---- pass 0: zero gt_max keys + counters ----------------------------------
__global__ void k0_init(uint32_t* __restrict__ ws_head) {
  int i = threadIdx.x;
  if (i < B_N * M_N + 2 * B_N) ws_head[i] = 0u;
}

// ---- pass 1: per-GT max IoU over all anchors ------------------------------
__global__ void k1_gtmax(const float4* __restrict__ anc,
                         const float4* __restrict__ gts,
                         uint32_t* __restrict__ gtmax_key) {
  __shared__ float4 sgt[M_N];
  __shared__ uint32_t skey[M_N];
  const int tid = threadIdx.x;
  const int b   = blockIdx.y;
  const int ai  = blockIdx.x * blockDim.x + tid;
  const int ac  = ai < A_N ? ai : A_N - 1;

  if (tid < M_N) {
    async_b128_to_lds((uint32_t)(uintptr_t)&sgt[tid], (const void*)(gts + b * M_N + tid));
    skey[tid] = 0u;
    wait_async0();
  }
  __builtin_prefetch((const void*)&anc[ac], 0, 3);   // global_prefetch_b8
  __syncthreads();

  float4 av = anc[ac];
  float area_a = (av.z - av.x) * (av.w - av.y);
  bool inimg = anchor_inimg(av);
  if (ai < A_N) {
#pragma unroll 8
    for (int m = 0; m < M_N; ++m) {
      uint32_t k = f2key(iou_masked(av, area_a, sgt[m], inimg));
      if (k > skey[m]) atomicMax(&skey[m], k);       // racy pre-check: kills contention
    }
  }
  __syncthreads();
  if (tid < M_N) atomicMax(&gtmax_key[b * M_N + tid], skey[tid]);
}

// ---- pass 2: conf_pre + pos/neg counts (WMMA as popcount reduction) -------
__global__ void k2_count(const float4* __restrict__ anc,
                         const float4* __restrict__ gts,
                         const uint32_t* __restrict__ gtmax_key,
                         int* __restrict__ counts,
                         int* __restrict__ conf_out) {
  __shared__ float4 sgt[M_N];
  __shared__ float sgmax[M_N];
  __shared__ int scnt[2];
  const int tid = threadIdx.x;
  const int b   = blockIdx.y;
  const int ai  = blockIdx.x * blockDim.x + tid;
  const int ac  = ai < A_N ? ai : A_N - 1;

  if (tid < M_N) {
    async_b128_to_lds((uint32_t)(uintptr_t)&sgt[tid], (const void*)(gts + b * M_N + tid));
    sgmax[tid] = key2f(gtmax_key[b * M_N + tid]);
    wait_async0();
  }
  if (tid < 2) scnt[tid] = 0;
  __builtin_prefetch((const void*)&anc[ac], 0, 3);
  __syncthreads();

  float4 av = anc[ac];
  float area_a = (av.z - av.x) * (av.w - av.y);
  bool inimg = anchor_inimg(av);
  float maxi = -1e30f;
  bool best = false;
#pragma unroll 8
  for (int m = 0; m < M_N; ++m) {
    float v = iou_masked(av, area_a, sgt[m], inimg);
    maxi = fmaxf(maxi, v);
    best |= (v == sgmax[m]) & (v >= 0.0f);
  }
  bool pos = (maxi >= POS_TH) | best;
  int conf = pos ? 1 : (maxi >= NEG_TH ? -1 : 0);
  if (!inimg) conf = -1;
  if (ai < A_N) conf_out[b * A_N + ai] = conf;

  // Wave32 popcount via matrix unit: plant count at A[m=lane%16, k=0|8], B=ones.
  // D[m][n] = val_m + val_{m+16}; sum c[0..7] gives a half-sum per 16-lane group,
  // one xor-16 swap completes the full 32-lane total. Uniform path: EXEC all-1s.
  int p = (ai < A_N) && (conf == 1);
  int n = (ai < A_N) && (conf == 0);
  v16h amp = {}, amn = {}, ones;
#pragma unroll
  for (int i = 0; i < 16; ++i) ones[i] = (_Float16)1.0f;
  amp[0] = (_Float16)(float)p;
  amn[0] = (_Float16)(float)n;
  v8f cp = {}, cn = {};
  cp = __builtin_amdgcn_wmma_f32_16x16x32_f16(false, amp, false, ones, (short)0, cp, false, false);
  cn = __builtin_amdgcn_wmma_f32_16x16x32_f16(false, amn, false, ones, (short)0, cn, false, false);
  float sp = 0.0f, sn = 0.0f;
#pragma unroll
  for (int i = 0; i < 8; ++i) { sp += cp[i]; sn += cn[i]; }
  sp += __shfl_xor(sp, 16, 32);
  sn += __shfl_xor(sn, 16, 32);
  if ((tid & 31) == 0) {
    atomicAdd(&scnt[0], (int)(sp + 0.5f));
    atomicAdd(&scnt[1], (int)(sn + 0.5f));
  }
  __syncthreads();
  if (tid == 0) {
    atomicAdd(&counts[2 * b + 0], scnt[0]);
    atomicAdd(&counts[2 * b + 1], scnt[1]);
  }
}

// ---- pass 3: subsample conf + regression offsets --------------------------
__global__ void k3_final(const float4* __restrict__ anc,
                         const float4* __restrict__ gts,
                         const int* __restrict__ counts,
                         int* __restrict__ conf_io,
                         float4* __restrict__ off_out) {
  __shared__ float4 sgt[M_N];
  const int tid = threadIdx.x;
  const int b   = blockIdx.y;
  const int ai  = blockIdx.x * blockDim.x + tid;
  const int ac  = ai < A_N ? ai : A_N - 1;

  if (tid < M_N) {
    async_b128_to_lds((uint32_t)(uintptr_t)&sgt[tid], (const void*)(gts + b * M_N + tid));
    wait_async0();
  }
  __builtin_prefetch((const void*)&anc[ac], 0, 3);
  __syncthreads();
  if (ai >= A_N) return;

  float4 av = anc[ai];
  float area_a = (av.z - av.x) * (av.w - av.y);
  bool inimg = anchor_inimg(av);
  float maxi = -1e30f;
  int arg = 0;
#pragma unroll 8
  for (int m = 0; m < M_N; ++m) {
    float v = iou_masked(av, area_a, sgt[m], inimg);
    if (v > maxi) { maxi = v; arg = m; }   // first-max, matches jnp.argmax
  }

  int conf = conf_io[b * A_N + ai];
  const int cpos = counts[2 * b + 0];
  const int cneg = counts[2 * b + 1];
  const int npos = min(cpos, MAX_POSK);
  const int nneg = min(cneg, MAX_SAMPLES - npos);
  uint32_t hp = hash32((uint32_t)ai * 0x9E3779B9u + (uint32_t)b * 0x85EBCA6Bu);
  if (conf == 1 && cpos > npos) {
    if ((hp % (uint32_t)cpos) >= (uint32_t)npos) conf = -1;
  }
  uint32_t hn = hash32(hp ^ 0x68BC21EBu);
  if (conf == 0 && cneg > nneg) {
    if ((hn % (uint32_t)cneg) >= (uint32_t)nneg) conf = -1;
  }
  conf_io[b * A_N + ai] = conf;

  int idx = (maxi >= POS_TH) ? arg : 0;
  float4 g = sgt[idx];
  float acx = (av.x + av.z) * 0.5f, acy = (av.y + av.w) * 0.5f;
  float aw  = av.z - av.x,          ah  = av.w - av.y;
  float tcx = (g.x + g.z) * 0.5f,   tcy = (g.y + g.w) * 0.5f;
  float tw  = g.z - g.x,            th  = g.w - g.y;
  float4 off;
  off.x = (tcx - acx) / aw;
  off.y = (tcy - acy) / ah;
  off.z = __logf(tw / aw);
  off.w = __logf(th / ah);
  if (!inimg) { off.x = 0.0f; off.y = 0.0f; off.z = 0.0f; off.w = 0.0f; }
  off_out[b * A_N + ai] = off;
}

extern "C" void kernel_launch(void* const* d_in, const int* in_sizes, int n_in,
                              void* d_out, int out_size, void* d_ws, size_t ws_size,
                              hipStream_t stream) {
  (void)in_sizes; (void)n_in; (void)out_size; (void)ws_size;
  // inputs: [0]=images (unused, shape only), [1]=in_anchor (A,4) f32,
  //         [2]=all_gt_bboxes (B,M,4) f32, [3]=all_gt_orig_classes (unused)
  const float4* anc = (const float4*)d_in[1];
  const float4* gts = (const float4*)d_in[2];
  uint32_t* gtkey = (uint32_t*)d_ws;                       // B*M keys
  int* counts     = (int*)d_ws + B_N * M_N;                // B*2 counters
  int* conf       = (int*)d_out;                           // (B,A) int32
  float4* off     = (float4*)((float*)d_out + (size_t)B_N * A_N);  // (B,A,4) f32

  dim3 grid((A_N + 255) / 256, B_N);
  k0_init <<<1, 512, 0, stream>>>(gtkey);
  k1_gtmax<<<grid, 256, 0, stream>>>(anc, gts, gtkey);
  k2_count<<<grid, 256, 0, stream>>>(anc, gts, gtkey, counts, conf);
  k3_final<<<grid, 256, 0, stream>>>(anc, gts, counts, conf, off);
}
